// SelfAttentionModel_58815282151499
// MI455X (gfx1250) — compile-verified, matrix operand order checked by reading, not run on hardware
//
#include <hip/hip_runtime.h>
#include <hip/hip_bf16.h>
#include <cstdint>

// ---------------- types / helpers ----------------
typedef __attribute__((ext_vector_type(16))) __bf16 v16bf;
typedef __attribute__((ext_vector_type(8)))  float  v8f;

static constexpr int  BSZ = 2, NSEQ = 16384, DIM = 512, NHEAD = 8, DH = 64;
static constexpr int  FFD = 2048, EXPRD = 512, DIFFD = 128, POSFD = 128;
static constexpr long NT = (long)BSZ * NSEQ;                 // 32768 tokens
static constexpr float QK_SCALE = 0.35355339059327373f;      // 64^-0.25

__device__ __forceinline__ uint16_t f2bf(float f) {
  uint32_t u = __builtin_bit_cast(uint32_t, f);
  uint32_t r = (u + 0x7FFFu + ((u >> 16) & 1u)) >> 16;       // RNE
  return (uint16_t)r;
}
__device__ __forceinline__ float bf2f(uint16_t h) {
  uint32_t u = ((uint32_t)h) << 16;
  return __builtin_bit_cast(float, u);
}

union FragBF { uint32_t u[8]; v16bf v; };

// CDNA5 async global->LDS copy (16B per lane, ASYNCcnt tracked).
// GV mode: lds byte offset in VGPR, 64-bit global address in VGPR pair.
__device__ __forceinline__ void async_b128(uint32_t lds_byte_off, uint64_t gaddr) {
  asm volatile("global_load_async_to_lds_b128 %0, %1, off"
               :: "v"(lds_byte_off), "v"(gaddr) : "memory");
}
__device__ __forceinline__ void wait_async0() {
  asm volatile("s_wait_asynccnt 0x0" ::: "memory");
}
// generic shared pointer -> LDS byte offset (ISA: LDS aperture maps addr[31:0])
__device__ __forceinline__ uint32_t lds_off(const void* p) {
  return (uint32_t)(uintptr_t)p;
}

// ---------------- generic tiled bf16 GEMM (WMMA f32_16x16x32_bf16) ----------
// C[M,N] = act( A[M,K](bf16) @ Wt[N,K](bf16)^T + bias[N] + addsrc[M,N](f32) )
// tile: 128x64, K step 32, double-buffered async global->LDS pipeline.
// 256 threads = 8 waves, wave w owns rows w*16..w*16+15.
#define SAP 20   // u32 stride per 32-wide bf16 row (16B aligned, conflict-free)
__global__ __launch_bounds__(256) void gemm_bf16_k(
    const uint16_t* __restrict__ A, int lda,
    const uint16_t* __restrict__ Wt, int ldw,
    const float* __restrict__ bias,
    const float* __restrict__ addsrc, int ldadd,
    void* __restrict__ out, int ldc,
    int K, int outBf, int act)
{
  __shared__ uint32_t sA[2][128 * SAP];
  __shared__ uint32_t sW[2][64 * SAP];
  const int tid  = threadIdx.x;
  const int lane = tid & 31;
  const int wave = tid >> 5;
  const long rowBase = (long)blockIdx.x * 128;
  const int  col0    = blockIdx.y * 64;

  v8f acc[4];
  #pragma unroll
  for (int c = 0; c < 4; ++c)
    #pragma unroll
    for (int r = 0; r < 8; ++r) acc[c][r] = 0.0f;

  // ---- async tile copy: A 128x32 bf16 (2 ops/lane), W 64x32 bf16 (1 op/lane)
  auto issue_tiles = [&](int k0, int buf) {
    #pragma unroll
    for (int i = 0; i < 2; ++i) {
      int ch = tid + i * 256;            // 512 16B chunks
      int row = ch >> 2, q = ch & 3;
      const uint16_t* gp = A + (rowBase + row) * lda + k0;
      async_b128(lds_off(&sA[buf][row * SAP + q * 4]),
                 (uint64_t)(uintptr_t)gp + (uint64_t)(q * 16));
    }
    {
      int row = tid >> 2, q = tid & 3;   // 256 16B chunks
      const uint16_t* gp = Wt + (long)(col0 + row) * ldw + k0;
      async_b128(lds_off(&sW[buf][row * SAP + q * 4]),
                 (uint64_t)(uintptr_t)gp + (uint64_t)(q * 16));
    }
  };

  issue_tiles(0, 0);
  wait_async0();
  __syncthreads();

  int buf = 0;
  for (int k0 = 0; k0 < K; k0 += 32) {
    const int nk = k0 + 32;
    if (nk < K) issue_tiles(nk, buf ^ 1);

    const int g = lane >> 4, mr = lane & 15;
    FragBF a;
    const uint32_t* ar = &sA[buf][(wave * 16 + mr) * SAP];
    #pragma unroll
    for (int j = 0; j < 4; ++j) { a.u[j] = ar[4 * g + j]; a.u[4 + j] = ar[8 + 4 * g + j]; }
    #pragma unroll
    for (int c = 0; c < 4; ++c) {
      FragBF b;
      const uint32_t* br = &sW[buf][(c * 16 + mr) * SAP];
      #pragma unroll
      for (int j = 0; j < 4; ++j) { b.u[j] = br[4 * g + j]; b.u[4 + j] = br[8 + 4 * g + j]; }
      acc[c] = __builtin_amdgcn_wmma_f32_16x16x32_bf16(false, a.v, false, b.v,
                                                       (short)0, acc[c], false, false);
    }
    if (nk < K) {
      wait_async0();       // own async copies into buf^1 complete
      __syncthreads();     // everyone's copies visible; prior reads of buf^1 done
      buf ^= 1;
    }
  }

  const int g = lane >> 4, nl = lane & 15;
  for (int c = 0; c < 4; ++c) {
    int n = col0 + c * 16 + nl;
    float bv = bias ? bias[n] : 0.0f;
    #pragma unroll
    for (int r = 0; r < 8; ++r) {
      long m = rowBase + wave * 16 + r + 8 * g;
      float v = acc[c][r] + bv;
      if (addsrc) v += addsrc[m * ldadd + n];
      if (act == 1)      v = v > 0.0f ? v : 0.0f;
      else if (act == 2) v = 0.5f * v * (1.0f + erff(v * 0.70710678118f)); // exact GELU
      if (outBf) ((uint16_t*)out)[m * ldc + n] = f2bf(v);
      else       ((float*)out)[m * ldc + n]    = v;
    }
  }
}

// ---------------- ctx = k^T v per (b,h), split-K partials (WMMA) -------------
__global__ __launch_bounds__(256) void ctx_k(
    const uint16_t* __restrict__ kq, const uint16_t* __restrict__ vv,
    float* __restrict__ ctxPart, int rowsPerSplit)
{
  __shared__ uint16_t skT[64 * 40];
  __shared__ uint16_t svT[64 * 40];
  const int bh = blockIdx.x, b = bh >> 3, h = bh & 7;
  const int n_start = blockIdx.y * rowsPerSplit;
  const int tid = threadIdx.x, lane = tid & 31, wave = tid >> 5;
  const int rb = wave >> 1, cbase = (wave & 1) * 2;

  v8f acc[2];
  #pragma unroll
  for (int c = 0; c < 2; ++c)
    #pragma unroll
    for (int r = 0; r < 8; ++r) acc[c][r] = 0.0f;

  for (int n0 = n_start; n0 < n_start + rowsPerSplit; n0 += 32) {
    #pragma unroll
    for (int i = 0; i < 4; ++i) {                 // 32 rows x 64 cols (as 32 u32/row)
      int e = tid + i * 256;
      int row = e >> 5, cp = e & 31;
      long base = ((long)b * NSEQ + n0 + row) * DIM + h * DH;
      uint32_t wk_ = ((const uint32_t*)(kq + base))[cp];
      skT[(cp * 2) * 40 + row]     = (uint16_t)wk_;
      skT[(cp * 2 + 1) * 40 + row] = (uint16_t)(wk_ >> 16);
      uint32_t wv_ = ((const uint32_t*)(vv + base))[cp];
      svT[(cp * 2) * 40 + row]     = (uint16_t)wv_;
      svT[(cp * 2 + 1) * 40 + row] = (uint16_t)(wv_ >> 16);
    }
    __syncthreads();
    const int g = lane >> 4, mr = lane & 15;
    FragBF a;
    const uint32_t* ar = (const uint32_t*)&skT[(rb * 16 + mr) * 40];
    #pragma unroll
    for (int j = 0; j < 4; ++j) { a.u[j] = ar[4 * g + j]; a.u[4 + j] = ar[8 + 4 * g + j]; }
    #pragma unroll
    for (int cc = 0; cc < 2; ++cc) {
      int cb = cbase + cc;
      FragBF bfr;
      const uint32_t* br = (const uint32_t*)&svT[(cb * 16 + mr) * 40];
      #pragma unroll
      for (int j = 0; j < 4; ++j) { bfr.u[j] = br[4 * g + j]; bfr.u[4 + j] = br[8 + 4 * g + j]; }
      acc[cc] = __builtin_amdgcn_wmma_f32_16x16x32_bf16(false, a.v, false, bfr.v,
                                                        (short)0, acc[cc], false, false);
    }
    __syncthreads();
  }
  const int g = lane >> 4, nl = lane & 15;
  float* dst = ctxPart + ((long)blockIdx.y * 16 + bh) * 4096;
  #pragma unroll
  for (int cc = 0; cc < 2; ++cc) {
    int cb = cbase + cc;
    #pragma unroll
    for (int r = 0; r < 8; ++r) {
      int m = rb * 16 + r + 8 * g;
      int n = cb * 16 + nl;
      dst[m * 64 + n] = acc[cc][r];
    }
  }
}

// reduce 8 split partials + transpose -> bf16 ctxT[bh][e][d]
__global__ void ctx_reduce_k(const float* __restrict__ ctxPart, uint16_t* __restrict__ ctxT) {
  int idx = blockIdx.x * 256 + threadIdx.x;      // 16*4096
  if (idx >= 16 * 4096) return;
  int bh = idx >> 12, r = idx & 4095, e = r >> 6, d = r & 63;
  float s = 0.0f;
  #pragma unroll
  for (int sp = 0; sp < 8; ++sp) s += ctxPart[((long)sp * 16 + bh) * 4096 + d * 64 + e];
  ctxT[idx] = f2bf(s);
}

// ---------------- LayerNorm (one wave per token) -> bf16 ---------------------
__global__ __launch_bounds__(256) void ln_k(const float* __restrict__ x,
                                            const float* __restrict__ gam,
                                            const float* __restrict__ bet,
                                            uint16_t* __restrict__ out, long ntok) {
  long tok = (long)blockIdx.x * 8 + (threadIdx.x >> 5);
  if (tok >= ntok) return;
  int lane = threadIdx.x & 31;
  const float* xr = x + tok * DIM;
  float vals[16], s = 0.0f, ss = 0.0f;
  #pragma unroll
  for (int i = 0; i < 16; ++i) { float v = xr[lane * 16 + i]; vals[i] = v; s += v; ss += v * v; }
  for (int o = 16; o > 0; o >>= 1) { s += __shfl_xor(s, o, 32); ss += __shfl_xor(ss, o, 32); }
  float mean = s * (1.0f / DIM);
  float var  = ss * (1.0f / DIM) - mean * mean;
  float rs   = rsqrtf(var + 1e-5f);
  #pragma unroll
  for (int i = 0; i < 16; ++i) {
    int c = lane * 16 + i;
    out[tok * DIM + c] = f2bf((vals[i] - mean) * rs * gam[c] + bet[c]);
  }
}

// ---------------- q softmax over head dim (wave per token-head, in place) ----
__global__ __launch_bounds__(256) void qsoft_k(uint16_t* __restrict__ q, long nth) {
  long idx = (long)blockIdx.x * 8 + (threadIdx.x >> 5);
  if (idx >= nth) return;
  int lane = threadIdx.x & 31;
  uint16_t* p = q + idx * 64;
  float v0 = bf2f(p[lane * 2]) * QK_SCALE, v1 = bf2f(p[lane * 2 + 1]) * QK_SCALE;
  float mx = fmaxf(v0, v1);
  for (int o = 16; o > 0; o >>= 1) mx = fmaxf(mx, __shfl_xor(mx, o, 32));
  float e0 = expf(v0 - mx), e1 = expf(v1 - mx);
  float s = e0 + e1;
  for (int o = 16; o > 0; o >>= 1) s += __shfl_xor(s, o, 32);
  float inv = 1.0f / s;
  p[lane * 2]     = f2bf(e0 * inv);
  p[lane * 2 + 1] = f2bf(e1 * inv);
}

// ---------------- k column stats (max, sumexp) over sequence -----------------
__global__ __launch_bounds__(256) void kstats_k(const uint16_t* __restrict__ k,
                                                float* __restrict__ kmax,
                                                float* __restrict__ ksum) {
  int col = blockIdx.x;                      // b*512 + d
  int b = col >> 9, d = col & 511;
  __shared__ float red[256];
  float mx = -1e30f;
  for (int n = threadIdx.x; n < NSEQ; n += 256)
    mx = fmaxf(mx, bf2f(k[((long)b * NSEQ + n) * DIM + d]) * QK_SCALE);
  red[threadIdx.x] = mx; __syncthreads();
  for (int o = 128; o > 0; o >>= 1) { if (threadIdx.x < o) red[threadIdx.x] = fmaxf(red[threadIdx.x], red[threadIdx.x + o]); __syncthreads(); }
  mx = red[0]; __syncthreads();
  float s = 0.0f;
  for (int n = threadIdx.x; n < NSEQ; n += 256)
    s += expf(bf2f(k[((long)b * NSEQ + n) * DIM + d]) * QK_SCALE - mx);
  red[threadIdx.x] = s; __syncthreads();
  for (int o = 128; o > 0; o >>= 1) { if (threadIdx.x < o) red[threadIdx.x] += red[threadIdx.x + o]; __syncthreads(); }
  if (threadIdx.x == 0) { kmax[col] = mx; ksum[col] = red[0]; }
}

__global__ void knorm_k(uint16_t* __restrict__ k, const float* __restrict__ kmax,
                        const float* __restrict__ ksum, long total) {
  long i = (long)blockIdx.x * 256 + threadIdx.x;
  if (i >= total) return;
  int d = (int)(i & 511);
  long tok = i >> 9;
  int b = (int)(tok / NSEQ);
  int col = b * 512 + d;
  float v = expf(bf2f(k[i]) * QK_SCALE - kmax[col]) / ksum[col];
  k[i] = f2bf(v);
}

// ---------------- small / elementwise kernels --------------------------------
__global__ void conv_bf16_k(const float* __restrict__ src, uint16_t* __restrict__ dst, long n) {
  long i = (long)blockIdx.x * 256 + threadIdx.x;
  if (i < n) dst[i] = f2bf(src[i]);
}

__global__ void transpose_conv_k(const float* __restrict__ src, int ldsrc, int K,
                                 uint16_t* __restrict__ dst, long total) {
  long i = (long)blockIdx.x * 256 + threadIdx.x;
  if (i >= total) return;
  long n = i / K; long k = i % K;
  dst[i] = f2bf(src[k * ldsrc + n]);
}

__global__ void initx_k(float* __restrict__ x, const float* __restrict__ tc,
                        long total, long nPerB) {
  long i = (long)blockIdx.x * 256 + threadIdx.x;
  if (i >= total) return;
  int b = (int)(i / nPerB);
  x[i] = tc[b * DIM + (int)(i & 511)];
}

// td = dt @ y_w + y_b  (write to d_out tail), time_contrib = c_b + td @ c_w[512:640]
__global__ __launch_bounds__(256) void prep_k(const float* __restrict__ dt,
    const float* __restrict__ yw, const float* __restrict__ yb,
    const float* __restrict__ cw, const float* __restrict__ cb,
    float* __restrict__ tc, float* __restrict__ td_out) {
  __shared__ float td_s[256];
  int t = threadIdx.x;
  int b = t >> 7, j = t & 127;
  float s = yb[j];
  for (int i = 0; i < DIFFD; ++i) s += dt[b * DIFFD + i] * yw[i * DIFFD + j];
  td_s[t] = s;
  td_out[t] = s;
  __syncthreads();
  for (int idx = t; idx < BSZ * DIM; idx += 256) {
    int bb = idx >> 9, jj = idx & 511;
    float acc = cb[jj];
    for (int i = 0; i < DIFFD; ++i) acc += td_s[bb * DIFFD + i] * cw[(512 + i) * DIM + jj];
    tc[idx] = acc;
  }
}

// position MLP: (3) -> dir+norm(4) -> 64 relu -> 128, bf16 out
__global__ __launch_bounds__(256) void pos_k(const float* __restrict__ pos,
    const float* __restrict__ w1, const float* __restrict__ b1,
    const float* __restrict__ w2, const float* __restrict__ b2,
    uint16_t* __restrict__ tp, long ntok) {
  long t = (long)blockIdx.x * 256 + threadIdx.x;
  if (t >= ntok) return;
  float p0 = pos[t * 3], p1 = pos[t * 3 + 1], p2 = pos[t * 3 + 2];
  float nr = sqrtf(p0 * p0 + p1 * p1 + p2 * p2);
  float inv = 1.0f / (nr + 1e-7f);
  float f0 = p0 * inv, f1 = p1 * inv, f2 = p2 * inv, f3 = nr;
  float h[64];
  #pragma unroll 8
  for (int j = 0; j < 64; ++j) {
    float s = b1[j] + f0 * w1[j] + f1 * w1[64 + j] + f2 * w1[128 + j] + f3 * w1[192 + j];
    h[j] = s > 0.0f ? s : 0.0f;
  }
  for (int o = 0; o < POSFD; ++o) {
    float s = b2[o];
    #pragma unroll 8
    for (int j = 0; j < 64; ++j) s += h[j] * w2[j * POSFD + o];
    tp[t * POSFD + o] = f2bf(s);
  }
}

// hp head: (512)->3, f32 x input, f32 out
__global__ __launch_bounds__(256) void hp_k(const float* __restrict__ x,
    const float* __restrict__ w, const float* __restrict__ b,
    float* __restrict__ out, long ntok) {
  long t = (long)blockIdx.x * 256 + threadIdx.x;
  if (t >= ntok) return;
  float s0 = b[0], s1 = b[1], s2 = b[2];
  const float* xr = x + t * DIM;
  for (int c = 0; c < DIM; ++c) {
    float v = xr[c];
    s0 += v * w[c * 3]; s1 += v * w[c * 3 + 1]; s2 += v * w[c * 3 + 2];
  }
  out[t * 3] = s0; out[t * 3 + 1] = s1; out[t * 3 + 2] = s2;
}

// ---------------- host launch -------------------------------------------------
extern "C" void kernel_launch(void* const* d_in, const int* in_sizes, int n_in,
                              void* d_out, int out_size, void* d_ws, size_t ws_size,
                              hipStream_t stream) {
  auto F = [&](int i) { return (const float*)d_in[i]; };
  const float* expr = F(0);  const float* dt   = F(1);  const float* pos = F(2);
  const float* p_w1 = F(3);  const float* p_b1 = F(4);  const float* p_w2 = F(5);  const float* p_b2 = F(6);
  const float* e_w1 = F(7);  const float* e_b1 = F(8);  const float* e_w2 = F(9);  const float* e_b2 = F(10);
  const float* c_w  = F(11); const float* c_b  = F(12);
  const float* y_w  = F(13); const float* y_b  = F(14);
  const float* ln1g = F(15); const float* ln1b = F(16);
  const float* wq   = F(17); const float* wk   = F(18); const float* wv = F(19);
  const float* wo   = F(20); const float* bo   = F(21);
  const float* ln2g = F(22); const float* ln2b = F(23);
  const float* f_w1 = F(24); const float* f_b1 = F(25);
  const float* f_w2 = F(26); const float* f_b2 = F(27);
  const float* hp_w = F(28); const float* hp_b = F(29);
  const float* he_w = F(30); const float* he_b = F(31);

  char* w = (char*)d_ws;
  size_t off = 0;
  auto alloc = [&](size_t bytes) { size_t r = off; off = (off + bytes + 255) & ~(size_t)255; return r; };

  uint16_t* wt_e1   = (uint16_t*)(w + alloc((size_t)64 * 512 * 2));
  uint16_t* wt_e2   = (uint16_t*)(w + alloc((size_t)512 * 64 * 2));
  uint16_t* wt_ctop = (uint16_t*)(w + alloc((size_t)512 * 512 * 2));
  uint16_t* wt_cpos = (uint16_t*)(w + alloc((size_t)512 * 128 * 2));
  uint16_t* wt_q    = (uint16_t*)(w + alloc((size_t)512 * 512 * 2));
  uint16_t* wt_k    = (uint16_t*)(w + alloc((size_t)512 * 512 * 2));
  uint16_t* wt_v    = (uint16_t*)(w + alloc((size_t)512 * 512 * 2));
  uint16_t* wt_o    = (uint16_t*)(w + alloc((size_t)512 * 512 * 2));
  uint16_t* wt_f1   = (uint16_t*)(w + alloc((size_t)2048 * 512 * 2));
  uint16_t* wt_f2   = (uint16_t*)(w + alloc((size_t)512 * 2048 * 2));
  uint16_t* wt_he   = (uint16_t*)(w + alloc((size_t)512 * 512 * 2));
  float*    tc      = (float*)   (w + alloc((size_t)BSZ * DIM * 4));
  float*    xres    = (float*)   (w + alloc((size_t)NT * DIM * 4));
  uint16_t* exprbf  = (uint16_t*)(w + alloc((size_t)NT * DIM * 2)); // also xbf
  uint16_t* hbf     = (uint16_t*)(w + alloc((size_t)NT * DIM * 2));
  uint16_t* te      = (uint16_t*)(w + alloc((size_t)NT * DIM * 2)); // also attn
  uint16_t* h1      = (uint16_t*)(w + alloc((size_t)NT * 64 * 2));
  uint16_t* tp      = (uint16_t*)(w + alloc((size_t)NT * POSFD * 2));
  uint16_t* qb      = (uint16_t*)(w + alloc((size_t)NT * DIM * 2));
  uint16_t* kb      = (uint16_t*)(w + alloc((size_t)NT * DIM * 2));
  uint16_t* vb      = (uint16_t*)(w + alloc((size_t)NT * DIM * 2));
  uint16_t* hid     = (uint16_t*)(w + alloc((size_t)NT * 512 * 2)); // FF chunk
  float*    kmax    = (float*)   (w + alloc((size_t)BSZ * DIM * 4));
  float*    ksum    = (float*)   (w + alloc((size_t)BSZ * DIM * 4));
  float*    ctxPart = (float*)   (w + alloc((size_t)8 * 16 * 4096 * 4));
  uint16_t* ctxT    = (uint16_t*)(w + alloc((size_t)16 * 4096 * 2));
  (void)ws_size; (void)n_in; (void)in_sizes; (void)out_size;

  auto tconv = [&](const float* src, int ldsrc, int K, int N, uint16_t* dst) {
    long total = (long)N * K;
    transpose_conv_k<<<(unsigned)((total + 255) / 256), 256, 0, stream>>>(src, ldsrc, K, dst, total);
  };
  auto gemm = [&](const uint16_t* Ap, int lda, const uint16_t* Wp, int ldw,
                  const float* bias, const float* addsrc, int ldadd,
                  void* out, int ldc, long M, int N, int K, int outBf, int act) {
    dim3 g((unsigned)(M / 128), (unsigned)(N / 64));
    gemm_bf16_k<<<g, 256, 0, stream>>>(Ap, lda, Wp, ldw, bias, addsrc, ldadd, out, ldc, K, outBf, act);
  };

  // 0) weight prep (f32 -> bf16, transposed to [N][K])
  tconv(e_w1, 64,   512, 64,   wt_e1);
  tconv(e_w2, 512,  64,  512,  wt_e2);
  tconv(c_w,  512,  512, 512,  wt_ctop);
  tconv(c_w + (size_t)640 * 512, 512, 128, 512, wt_cpos);
  tconv(wq,   512,  512, 512,  wt_q);
  tconv(wk,   512,  512, 512,  wt_k);
  tconv(wv,   512,  512, 512,  wt_v);
  tconv(wo,   512,  512, 512,  wt_o);
  tconv(f_w1, 2048, 512, 2048, wt_f1);
  tconv(f_w2, 512,  2048, 512, wt_f2);
  tconv(he_w, 512,  512, 512,  wt_he);

  // 1) td + time contribution; td -> d_out tail
  long he_elems = NT * DIM, hp_elems = NT * 3;
  float* td_out = (float*)d_out + he_elems + hp_elems;
  prep_k<<<1, 256, 0, stream>>>(dt, y_w, y_b, c_w, c_b, tc, td_out);

  // 2) expression MLP: h1 = relu(expr@e_w1 + b), te = h1@e_w2 + b
  conv_bf16_k<<<(unsigned)((NT * DIM + 255) / 256), 256, 0, stream>>>(expr, exprbf, NT * DIM);
  gemm(exprbf, 512, wt_e1, 512, e_b1, nullptr, 0, h1, 64,  NT, 64,  512, 1, 1);
  gemm(h1,     64,  wt_e2, 64,  e_b2, nullptr, 0, te, 512, NT, 512, 64,  1, 0);

  // 3) x = te@c_w_top + tp@c_w_pos + (c_b + time_contrib)
  initx_k<<<(unsigned)((NT * DIM + 255) / 256), 256, 0, stream>>>(xres, tc, NT * DIM, (long)NSEQ * DIM);
  gemm(te, 512, wt_ctop, 512, nullptr, xres, 512, xres, 512, NT, 512, 512, 0, 0);
  pos_k<<<(unsigned)((NT + 255) / 256), 256, 0, stream>>>(pos, p_w1, p_b1, p_w2, p_b2, tp, NT);
  gemm(tp, 128, wt_cpos, 128, nullptr, xres, 512, xres, 512, NT, 512, 128, 0, 0);

  // 4) linear attention
  ln_k<<<(unsigned)(NT / 8), 256, 0, stream>>>(xres, ln1g, ln1b, hbf, NT);
  gemm(hbf, 512, wt_q, 512, nullptr, nullptr, 0, qb, 512, NT, 512, 512, 1, 0);
  gemm(hbf, 512, wt_k, 512, nullptr, nullptr, 0, kb, 512, NT, 512, 512, 1, 0);
  gemm(hbf, 512, wt_v, 512, nullptr, nullptr, 0, vb, 512, NT, 512, 512, 1, 0);
  qsoft_k<<<(unsigned)(NT * NHEAD / 8), 256, 0, stream>>>(qb, NT * NHEAD);
  kstats_k<<<(unsigned)(BSZ * DIM), 256, 0, stream>>>(kb, kmax, ksum);
  knorm_k<<<(unsigned)((NT * DIM + 255) / 256), 256, 0, stream>>>(kb, kmax, ksum, NT * DIM);
  ctx_k<<<dim3(16, 8), 256, 0, stream>>>(kb, vb, ctxPart, NSEQ / 8);
  ctx_reduce_k<<<(unsigned)((16 * 4096 + 255) / 256), 256, 0, stream>>>(ctxPart, ctxT);
  for (int b = 0; b < BSZ; ++b)
    for (int h = 0; h < NHEAD; ++h) {
      long boff = (long)b * NSEQ * DIM + h * DH;
      gemm(qb + boff, 512, ctxT + (size_t)(b * NHEAD + h) * 4096, 64,
           nullptr, nullptr, 0, te + boff, 512, NSEQ, 64, 64, 1, 0);
    }
  gemm(te, 512, wt_o, 512, bo, xres, 512, xres, 512, NT, 512, 512, 0, 0);

  // 5) feedforward (chunked over the 2048 hidden to bound workspace)
  ln_k<<<(unsigned)(NT / 8), 256, 0, stream>>>(xres, ln2g, ln2b, hbf, NT);
  for (int c = 0; c < 4; ++c) {
    gemm(hbf, 512, wt_f1 + (size_t)c * 512 * 512, 512, f_b1 + c * 512,
         nullptr, 0, hid, 512, NT, 512, 512, 1, 2);
    gemm(hid, 512, wt_f2 + (size_t)c * 512, 2048, (c == 0 ? f_b2 : nullptr),
         xres, 512, xres, 512, NT, 512, 512, 0, 0);
  }

  // 6) output heads
  conv_bf16_k<<<(unsigned)((NT * DIM + 255) / 256), 256, 0, stream>>>(xres, exprbf, NT * DIM);
  gemm(exprbf, 512, wt_he, 512, he_b, nullptr, 0, (float*)d_out, 512, NT, 512, 512, 0, 0);
  hp_k<<<(unsigned)((NT + 255) / 256), 256, 0, stream>>>(xres, hp_w, hp_b,
                                                         (float*)d_out + he_elems, NT);
}
